// VLSTMCell_79594333929637
// MI455X (gfx1250) — compile-verified, matrix operand order checked by reading, not run on hardware
//
#include <hip/hip_runtime.h>
#include <hip/hip_bf16.h>
#include <math.h>

// ---------------------------------------------------------------------------
// VLSTM cell for MI455X (gfx1250, wave32).
//   x = [inputs | h_tm1 | z_tm1] @ [kernel ; recurrent_kernel ; recurrent_latent] + bias
//   gates -> h.  bf16 WMMA (v_wmma_f32_16x16x32_bf16), fp32 accumulate.
//
// Two-phase plan (when ws_size allows, 151 MB):
//   phase 1: convert fp32 -> bf16 ONCE, packed in WMMA-fragment-native layout.
//   phase 2: GEMM with zero conversion / zero transpose:
//            A fragments: direct coalesced global_load_b128 (software-pipelined),
//            B tiles: ASYNC global->LDS DMA (double-buffered, ASYNCcnt
//            software pipeline); B fragments double-buffered in registers.
// Fallback (small ws): single fused kernel converting on the fly.
// ---------------------------------------------------------------------------

#define BATCH   4096
#define UNITS   2048
#define INDIM   2048
#define NWIDE   (4 * UNITS)     // 8192

#define BM      128             // block rows
#define BNH     32              // h columns per block -> 128 x-columns (4 gates)

typedef __attribute__((ext_vector_type(16))) __bf16 v16bf;
typedef __attribute__((ext_vector_type(8)))  float  v8f;
typedef __attribute__((ext_vector_type(4)))  float  f32x4;
typedef __attribute__((ext_vector_type(4)))  unsigned int u32x4;

union FragBF16 {
    v16bf v;
    u32x4 q[2];
};

__device__ __forceinline__ unsigned short bf16_bits(float f) {
    unsigned u = __builtin_bit_cast(unsigned, f);
    u += 0x7FFFu + ((u >> 16) & 1u);          // round-to-nearest-even
    return (unsigned short)(u >> 16);
}

__device__ __forceinline__ unsigned pack2(float a, float b) {
    return (unsigned)bf16_bits(a) | ((unsigned)bf16_bits(b) << 16);
}

__device__ __forceinline__ float hsig(float x) {
    return fminf(fmaxf(fmaf(0.2f, x, 0.5f), 0.0f), 1.0f);
}

// async 16B global->LDS copy; offset: applies to BOTH lds and global address
#define ASYNC_CP16(lds32, gptr, OFF) \
    asm volatile("global_load_async_to_lds_b128 %0, %1, off offset:" OFF \
                 :: "v"(lds32), "v"(gptr) : "memory")

// ===========================================================================
// Phase 1a: pack activations [inputs|h|z] (M=4096,K=2048 each) into fragment
// layout. chunk id == global thread id -> fully coalesced 16B stores.
//   gid = ((s*256 + mTile)*64 + kStep)*64 + lane*2 + q
//   lane = (m&15) + 16*((K'>>3)&1),  q = K'>>4,  K' = k - kStep*32
// ===========================================================================
__global__ __launch_bounds__(256)
void pack_a_kernel(const float* __restrict__ inputs,
                   const float* __restrict__ h_tm1,
                   const float* __restrict__ z_tm1,
                   unsigned short* __restrict__ pA) {
    const unsigned gid  = blockIdx.x * 256u + threadIdx.x;
    const unsigned cq    = gid & 1u;
    const unsigned lane  = (gid >> 1) & 31u;
    const unsigned kStep = (gid >> 6) & 63u;
    const unsigned mTile = (gid >> 12) & 255u;
    const unsigned s     = gid >> 20;          // 0..2
    const float* src = (s == 0) ? inputs : (s == 1) ? h_tm1 : z_tm1;
    const unsigned m     = mTile * 16u + (lane & 15u);
    const unsigned kbase = kStep * 32u + (lane >> 4) * 8u + cq * 16u;
    const float* p = src + (size_t)m * INDIM + kbase;
    f32x4 f0 = *reinterpret_cast<const f32x4*>(p);
    f32x4 f1 = *reinterpret_cast<const f32x4*>(p + 4);
    u32x4 o;
    o.x = pack2(f0.x, f0.y);
    o.y = pack2(f0.z, f0.w);
    o.z = pack2(f1.x, f1.y);
    o.w = pack2(f1.z, f1.w);
    *reinterpret_cast<u32x4*>(pA + (size_t)gid * 8u) = o;
}

// ===========================================================================
// Phase 1b: pack weights (K=2048, N=8192 each, row-major) into B-fragment
// layout (lane = column, K striped). One thread = one 16B chunk.
//   gid = ((s*512 + nTile)*64 + kStep)*64 + lane*2 + q
// ===========================================================================
__global__ __launch_bounds__(256)
void pack_b_kernel(const float* __restrict__ Wk,
                   const float* __restrict__ Wr,
                   const float* __restrict__ Wl,
                   unsigned short* __restrict__ pB) {
    const unsigned gid  = blockIdx.x * 256u + threadIdx.x;
    const unsigned cq    = gid & 1u;
    const unsigned lane  = (gid >> 1) & 31u;
    const unsigned kStep = (gid >> 6) & 63u;
    const unsigned nTile = (gid >> 12) & 511u;
    const unsigned s     = gid >> 21;          // 0..2
    const float* src = (s == 0) ? Wk : (s == 1) ? Wr : Wl;
    const unsigned n     = nTile * 16u + (lane & 15u);
    const unsigned kbase = kStep * 32u + (lane >> 4) * 8u + cq * 16u;
    const float* p = src + (size_t)kbase * NWIDE + n;
    unsigned short v[8];
#pragma unroll
    for (int j = 0; j < 8; ++j) v[j] = bf16_bits(p[(size_t)j * NWIDE]);
    u32x4 o;
    o.x = (unsigned)v[0] | ((unsigned)v[1] << 16);
    o.y = (unsigned)v[2] | ((unsigned)v[3] << 16);
    o.z = (unsigned)v[4] | ((unsigned)v[5] << 16);
    o.w = (unsigned)v[6] | ((unsigned)v[7] << 16);
    *reinterpret_cast<u32x4*>(pB + (size_t)gid * 8u) = o;
}

// ===========================================================================
// Phase 2: GEMM + fused gate epilogue. 256 thr = 8 waves, each wave owns
// 16 rows x all 128 x-columns (4 gates x 32) -> gate math stays in-wave.
// 48 stages (3 srcs x 16), each = 4 k-steps = 32 WMMAs/wave.
// B staging: async DMA into double-buffered LDS, pipelined one stage ahead.
// A/B fragments: double-buffered registers (loads issued one WMMA ahead).
// ===========================================================================
__global__ __launch_bounds__(256)
void vlstm_wmma_packed(const unsigned short* __restrict__ pA,
                       const unsigned short* __restrict__ pB,
                       const float* __restrict__ c_tm1,
                       const float* __restrict__ bias,
                       float* __restrict__ out) {
    // 2 buffers x (8 frag regions x 4 k-steps x 1KB) = 2 x 32KB
    __shared__ alignas(16) unsigned short Bs[2][8 * 4 * 512];

    const int t    = threadIdx.x;
    const int lane = t & 31;
    const int wave = t >> 5;             // 0..7 == B staging region
    const int lidx = lane & 15;
    const int khalf = lane >> 4;
    const int m0t  = blockIdx.y * (BM / 16);   // mTile base (8 tiles/block)
    const int n0   = blockIdx.x * BNH;
    const int n0t  = n0 >> 4;
    const int gate    = wave >> 1;
    const int tilesub = wave & 1;

    // LDS byte address of this thread's staging slot in buffer 0
    // (low 32 bits of a flat shared-aperture address == LDS byte address)
    const unsigned lds0 = (unsigned)(size_t)(&Bs[0][wave * 2048 + lane * 64]);

    const v8f vzero = {0.f, 0.f, 0.f, 0.f, 0.f, 0.f, 0.f, 0.f};
    v8f acc[8];
#pragma unroll
    for (int f = 0; f < 8; ++f) acc[f] = vzero;

    // issue one stage (8 async b128 = 128B per thread, 32KB per block)
    auto issue_stage = [&](int stg) {
        const int s   = stg >> 4;
        const int ks0 = (stg & 15) * 4;
        const size_t bBase =
            ((size_t)(s * 512 + gate * 128 + n0t + tilesub) * 64u + ks0) * 512u;
        const unsigned short* g = pB + bBase + lane * 64;
        const unsigned l = lds0 + (unsigned)((stg & 1) * 32768);
        ASYNC_CP16(l, g, "0");
        ASYNC_CP16(l, g, "16");
        ASYNC_CP16(l, g, "32");
        ASYNC_CP16(l, g, "48");
        ASYNC_CP16(l, g, "64");
        ASYNC_CP16(l, g, "80");
        ASYNC_CP16(l, g, "96");
        ASYNC_CP16(l, g, "112");
    };

    issue_stage(0);

#pragma unroll 1
    for (int stg = 0; stg < 48; ++stg) {
        if (stg < 47) {
            issue_stage(stg + 1);   // pipeline next stage behind compute
            asm volatile("s_wait_asynccnt 0x8" ::: "memory");  // prev stage landed
        } else {
            asm volatile("s_wait_asynccnt 0x0" ::: "memory");
        }
        __syncthreads();

        const int s   = stg >> 4;
        const int ks0 = (stg & 15) * 4;
        const unsigned short* aPtr =
            pA + ((size_t)(s * 256 + m0t + wave) * 64u + ks0) * 512u + lane * 16;
        const unsigned short* bPtr =
            &Bs[stg & 1][0] + lane * 16;

        // -------- register double-buffer pipeline: A over kl, B over f -----
        FragBF16 a0, a1, b0, b1;
        {   // preload A(kl=0) and B(kl=0,f=0)
            const u32x4* ap = reinterpret_cast<const u32x4*>(aPtr);
            a0.q[0] = ap[0]; a0.q[1] = ap[1];
            const u32x4* bp = reinterpret_cast<const u32x4*>(bPtr);
            b0.q[0] = bp[0]; b0.q[1] = bp[1];
        }
#pragma unroll
        for (int kl = 0; kl < 4; ++kl) {
            FragBF16& acur = (kl & 1) ? a1 : a0;
            FragBF16& anxt = (kl & 1) ? a0 : a1;
            if (kl < 3) {   // prefetch next k-step's A behind this burst
                const u32x4* ap = reinterpret_cast<const u32x4*>(
                    aPtr + (size_t)(kl + 1) * 512u);
                anxt.q[0] = ap[0]; anxt.q[1] = ap[1];
            }
#pragma unroll
            for (int f = 0; f < 8; ++f) {
                FragBF16& bcur = (f & 1) ? b1 : b0;
                FragBF16& bnxt = (f & 1) ? b0 : b1;
                // issue loads for the NEXT fragment before consuming current
                if (f < 7) {
                    const u32x4* bp = reinterpret_cast<const u32x4*>(
                        bPtr + (f + 1) * 2048 + kl * 512);
                    bnxt.q[0] = bp[0]; bnxt.q[1] = bp[1];
                } else if (kl < 3) {
                    const u32x4* bp = reinterpret_cast<const u32x4*>(
                        bPtr + (kl + 1) * 512);
                    bnxt.q[0] = bp[0]; bnxt.q[1] = bp[1];
                }
                acc[f] = __builtin_amdgcn_wmma_f32_16x16x32_bf16(
                    false, acur.v, false, bcur.v,
                    (short)0, acc[f], false, false);
            }
        }
        __syncthreads();   // all waves done reading buf before it is re-filled
    }

    // ---- epilogue: bias + gates (C/D: lane 0..15 -> N, vgpr -> M row) ----
    const int rowBase = blockIdx.y * BM + wave * 16 + khalf * 8;
#pragma unroll
    for (int j = 0; j < 2; ++j) {
        const int ng = n0 + j * 16 + lidx;
        const float bi = bias[ng];
        const float bf = bias[UNITS + ng];
        const float bc = bias[2 * UNITS + ng];
        const float bo = bias[3 * UNITS + ng];
        const v8f xi = acc[0 + j];
        const v8f xf = acc[2 + j];
        const v8f xc = acc[4 + j];
        const v8f xo = acc[6 + j];
#pragma unroll
        for (int r = 0; r < 8; ++r) {
            const size_t idx = (size_t)(rowBase + r) * UNITS + ng;
            const float cprev = c_tm1[idx];
            const float gi = hsig(xi[r] + bi);
            const float gf = hsig(xf[r] + bf);
            const float cn = gf * cprev + gi * tanhf(xc[r] + bc);
            const float go = hsig(xo[r] + bo);
            out[idx] = go * tanhf(cn);
        }
    }
}

// ===========================================================================
// Fallback: fused on-the-fly conversion kernel (used only if ws too small).
// ===========================================================================
#define LDA 72
#define LDB 72

__global__ __launch_bounds__(256)
void vlstm_wmma_fused(const float* __restrict__ inputs,
                      const float* __restrict__ h_tm1,
                      const float* __restrict__ c_tm1,
                      const float* __restrict__ z_tm1,
                      const float* __restrict__ Wk,
                      const float* __restrict__ Wr,
                      const float* __restrict__ Wl,
                      const float* __restrict__ bias,
                      float* __restrict__ out) {
    __shared__ alignas(16) unsigned short As[BM * LDA];
    __shared__ alignas(16) unsigned short Bt[4 * BNH * LDB];

    const int t    = threadIdx.x;
    const int m0   = blockIdx.y * BM;
    const int n0   = blockIdx.x * BNH;
    const int lane  = t & 31;
    const int wave  = t >> 5;
    const int lidx  = lane & 15;
    const int khalf = lane >> 4;
    const int arow = t >> 1;
    const int acol = (t & 1) * 32;
    const int bk = t & 63;
    const int bg = t >> 6;

    const v8f vzero = {0.f, 0.f, 0.f, 0.f, 0.f, 0.f, 0.f, 0.f};
    v8f acc[8];
#pragma unroll
    for (int f = 0; f < 8; ++f) acc[f] = vzero;

#pragma unroll 1
    for (int s = 0; s < 3; ++s) {
        const float* Asrc = (s == 0) ? inputs : (s == 1) ? h_tm1 : z_tm1;
        const float* Bsrc = (s == 0) ? Wk     : (s == 1) ? Wr    : Wl;
#pragma unroll 1
        for (int k0 = 0; k0 < INDIM; k0 += 64) {
            {
                const float* ag = Asrc + (size_t)(m0 + arow) * INDIM + k0 + acol;
                unsigned* as32 = reinterpret_cast<unsigned*>(&As[arow * LDA + acol]);
#pragma unroll
                for (int c = 0; c < 32; c += 4) {
                    f32x4 f = *reinterpret_cast<const f32x4*>(ag + c);
                    as32[(c >> 1) + 0] = pack2(f.x, f.y);
                    as32[(c >> 1) + 1] = pack2(f.z, f.w);
                }
            }
            {
                const float* brow = Bsrc + (size_t)(k0 + bk) * NWIDE + n0;
#pragma unroll
                for (int hg = 0; hg < 2; ++hg) {
                    const int g16 = bg * 2 + hg;
                    const float* src = brow + (size_t)(g16 >> 1) * UNITS + (g16 & 1) * 16;
                    const int colTile = g16 * 16;
#pragma unroll
                    for (int c = 0; c < 16; c += 4) {
                        f32x4 f = *reinterpret_cast<const f32x4*>(src + c);
                        Bt[(colTile + c + 0) * LDB + bk] = bf16_bits(f.x);
                        Bt[(colTile + c + 1) * LDB + bk] = bf16_bits(f.y);
                        Bt[(colTile + c + 2) * LDB + bk] = bf16_bits(f.z);
                        Bt[(colTile + c + 3) * LDB + bk] = bf16_bits(f.w);
                    }
                }
            }
            __syncthreads();
#pragma unroll
            for (int ks = 0; ks < 64; ks += 32) {
                FragBF16 af;
                const unsigned short* ab = &As[(wave * 16 + lidx) * LDA + ks + khalf * 8];
                af.q[0] = *reinterpret_cast<const u32x4*>(ab);
                af.q[1] = *reinterpret_cast<const u32x4*>(ab + 16);
#pragma unroll
                for (int f = 0; f < 8; ++f) {
                    FragBF16 bf_;
                    const unsigned short* bb = &Bt[(f * 16 + lidx) * LDB + ks + khalf * 8];
                    bf_.q[0] = *reinterpret_cast<const u32x4*>(bb);
                    bf_.q[1] = *reinterpret_cast<const u32x4*>(bb + 16);
                    acc[f] = __builtin_amdgcn_wmma_f32_16x16x32_bf16(
                        false, af.v, false, bf_.v, (short)0, acc[f], false, false);
                }
            }
            __syncthreads();
        }
    }

    const int rowBase = m0 + wave * 16 + khalf * 8;
#pragma unroll
    for (int j = 0; j < 2; ++j) {
        const int ng = n0 + j * 16 + lidx;
        const float bi = bias[ng];
        const float bf = bias[UNITS + ng];
        const float bc = bias[2 * UNITS + ng];
        const float bo = bias[3 * UNITS + ng];
        const v8f xi = acc[0 + j];
        const v8f xf = acc[2 + j];
        const v8f xc = acc[4 + j];
        const v8f xo = acc[6 + j];
#pragma unroll
        for (int r = 0; r < 8; ++r) {
            const size_t idx = (size_t)(rowBase + r) * UNITS + ng;
            const float cprev = c_tm1[idx];
            const float gi = hsig(xi[r] + bi);
            const float gf = hsig(xf[r] + bf);
            const float cn = gf * cprev + gi * tanhf(xc[r] + bc);
            const float go = hsig(xo[r] + bo);
            out[idx] = go * tanhf(cn);
        }
    }
}

// ===========================================================================
extern "C" void kernel_launch(void* const* d_in, const int* in_sizes, int n_in,
                              void* d_out, int out_size, void* d_ws, size_t ws_size,
                              hipStream_t stream) {
    const float* inputs = (const float*)d_in[0];
    const float* h_tm1  = (const float*)d_in[1];
    const float* c_tm1  = (const float*)d_in[2];
    const float* z_tm1  = (const float*)d_in[3];
    const float* Wk     = (const float*)d_in[4];
    const float* Wr     = (const float*)d_in[5];
    const float* Wl     = (const float*)d_in[6];
    const float* bias   = (const float*)d_in[7];
    float* out = (float*)d_out;

    const size_t nA = 3ull * BATCH * INDIM;   // bf16 elements
    const size_t nB = 3ull * INDIM * NWIDE;
    const size_t need = (nA + nB) * sizeof(unsigned short);   // ~151 MB

    dim3 grid(UNITS / BNH, BATCH / BM);   // (64, 32)

    if (ws_size >= need) {
        unsigned short* pA = (unsigned short*)d_ws;
        unsigned short* pB = pA + nA;
        pack_a_kernel<<<(unsigned)(nA / 8 / 256), 256, 0, stream>>>(inputs, h_tm1, z_tm1, pA);
        pack_b_kernel<<<(unsigned)(nB / 8 / 256), 256, 0, stream>>>(Wk, Wr, Wl, pB);
        vlstm_wmma_packed<<<grid, 256, 0, stream>>>(pA, pB, c_tm1, bias, out);
    } else {
        vlstm_wmma_fused<<<grid, 256, 0, stream>>>(
            inputs, h_tm1, c_tm1, z_tm1, Wk, Wr, Wl, bias, out);
    }
}
